// VQVAE_16458314678629
// MI455X (gfx1250) — matrix operand order repaired and checked by reading, not compiled, not run on hardware
//
#include <hip/hip_runtime.h>
#include <vector>
#include <cmath>

// ---------------------------------------------------------------------------
// CDNA5 / gfx1250 VQ-VAE forward.
// All convs are LDS-tiled implicit GEMMs on v_wmma_f32_16x16x32_bf16
// (bf16 in, f32 accumulate). Conv weight tiles are staged global->LDS by the
// Tensor Data Mover (tensor_load_to_lds + s_wait_tensorcnt) when the builtin
// is available; activations are im2col-gathered to LDS as bf16. Attention
// QK^T and V*P also run on WMMA with the score strip staged in LDS.
// Fragment layouts per the CDNA5 ISA:
//   A 16x32 bf16: lanes 0-15 hold K{0..7,16..23}, lanes 16-31 K{8..15,24..31}
//   B 32x16 bf16: same k split as A; this code feeds A and B K 16h..16h+15 in lane half h
//   C/D 16x16 f32: VGPR r -> row r (lanes 0-15) / r+8 (lanes 16-31)
// ---------------------------------------------------------------------------

typedef __attribute__((ext_vector_type(16))) __bf16 v16bf;
typedef __attribute__((ext_vector_type(8)))  float  v8f;
typedef unsigned int u32x4 __attribute__((ext_vector_type(4)));
typedef int          i32x4 __attribute__((ext_vector_type(4)));
typedef int          i32x8 __attribute__((ext_vector_type(8)));

#if __has_builtin(__builtin_amdgcn_tensor_load_to_lds) && \
    __has_builtin(__builtin_amdgcn_s_wait_tensorcnt)
#define HAVE_TDM 1
#else
#define HAVE_TDM 0
#endif

union FragU  { v16bf v; float4 q[2]; };
union Ld16U  { float4 q[4]; float f[16]; };

__device__ __forceinline__ __bf16 f2bf(float f) {
  unsigned u = __builtin_bit_cast(unsigned, f);
  unsigned r = u + 0x7FFFu + ((u >> 16) & 1u);   // round-to-nearest-even
  return __builtin_bit_cast(__bf16, (unsigned short)(r >> 16));
}

// ---------------------------------------------------------------------------
// Implicit-GEMM conv (NCHW in, OIHW weights). M = N*Ho*Wo, Ncol = Cout,
// K = Cin*KH*KW. Block tile 128x64, K-chunk 32. 256 threads = 8 waves
// (4 along M x 2 along N); each wave owns 32x32 -> 4 WMMAs per K-chunk.
// ---------------------------------------------------------------------------
#define BM 128
#define BN 64
#define BK 32
#define LDA 48     // A row stride in halves (16B aligned, padded)
#define LDB 48     // fallback B row stride in halves
#define LDBF 36    // TDM B row stride in floats (32 data + 4 pad DWORDs)

__global__ __launch_bounds__(256)
void vq_conv_wmma(const float* __restrict__ X, const float* __restrict__ Wt,
                  const float* __restrict__ Bs, float* __restrict__ Y,
                  int N, int Cin, int H, int W,
                  int Cout, int Ho, int Wo,
                  int KH, int KW, int stride, int pad)
{
  __shared__ __align__(16) __bf16 As[BM * LDA];
#if HAVE_TDM
  __shared__ __align__(16) float  Bf[BN * LDBF];   // f32 tile written by TDM
#else
  __shared__ __align__(16) __bf16 Bsh[BN * LDB];
#endif

  const int t    = threadIdx.x;
  const int lane = t & 31, wave = t >> 5;
  const int half = lane >> 4, lm = lane & 15;
  const int waveM = wave >> 1, waveN = wave & 1;    // 4 x 2 wave grid
  const int Mtot = N * Ho * Wo;
  const int K    = Cin * KH * KW;
  const int KHW  = KH * KW;
  const int m0   = blockIdx.x * BM;
  const int n0   = blockIdx.y * BN;

  // ---- A-staging: this thread's fixed pixel row (2 threads per row) ----
  const int sm    = t & 127;         // row of A tile
  const int sksel = t >> 7;          // which k of each pair
  int  mA  = m0 + sm;
  bool mok = (mA < Mtot);
  int  mc  = mok ? mA : 0;
  int  wo_s = mc % Wo;  mc /= Wo;
  int  ho_s = mc % Ho;
  int  nb_s = mc / Ho;

#if !HAVE_TDM
  // ---- fallback B-staging: 4 threads per cout row, 8 consecutive k ----
  const int sn  = t >> 2;
  const int skb = (t & 3) * 8;
  const bool nok_s = (n0 + sn) < Cout;
  const float* wrow_s = Wt + (size_t)(nok_s ? (n0 + sn) : 0) * K;
#endif

  v8f acc[2][2] = {};

  for (int k0 = 0; k0 < K; k0 += BK) {
#if HAVE_TDM
    // ---- B tile via Tensor Data Mover: [64 rows x 32 f32], row stride K,
    //      zero-fill past (K - k0) cols / (Cout - n0) rows, LDS rows padded
    //      to 36 floats (pad_interval = 32 DWORDs, pad_amount = 4 DWORDs) ----
    if (wave == 0) {
      size_t  ga   = (size_t)(Wt + (size_t)n0 * K + k0);     // byte address
      unsigned lds = (unsigned)(size_t)(void*)Bf;            // LDS byte offset
      unsigned dim0 = (unsigned)(K - k0);
      unsigned dim1 = (unsigned)(Cout > n0 ? Cout - n0 : 0);
      unsigned long long s0 = (unsigned long long)K;         // dim0 stride
      u32x4 g0 = {};
      g0[0] = 1u;                                            // count=1, load
      g0[1] = lds;                                           // lds_addr
      g0[2] = (unsigned)(ga & 0xFFFFFFFFull);                // global_addr lo
      g0[3] = (unsigned)((ga >> 32) & 0x1FFFFFFull) | (2u << 30);  // hi | type=2
      i32x8 g1 = {};
      g1[0] = (int)((2u << 16) | (1u << 20) | (4u << 22) | (3u << 25));
      g1[1] = (int)((dim0 & 0xFFFFu) << 16);                 // tensor_dim0 lo16
      g1[2] = (int)((dim0 >> 16) | ((dim1 & 0xFFFFu) << 16));
      g1[3] = (int)((dim1 >> 16) | (32u << 16));             // tile_dim0 = 32
      g1[4] = (int)(64u);                                    // tile_dim1 = 64
      g1[5] = (int)(unsigned)(s0 & 0xFFFFFFFFull);           // stride0 lo32
      g1[6] = (int)(unsigned)((s0 >> 32) & 0xFFFFull);       // stride0 hi16
      g1[7] = 0;                                             // stride1 = 0
      i32x4 gz = {};
#if __clang_major__ >= 23
      i32x8 gz8 = {};
      __builtin_amdgcn_tensor_load_to_lds(g0, g1, gz, gz, gz8, 0);
#else
      __builtin_amdgcn_tensor_load_to_lds(g0, g1, gz, gz, 0);
#endif
    }
#endif

    // ---- stage A tile (im2col gather, f32 -> bf16), overlaps the TDM ----
#pragma unroll
    for (int kk = 0; kk < BK; kk += 2) {
      int k = k0 + kk + sksel;
      float av = 0.0f;
      if (mok && k < K) {
        int ci = k / KHW;
        int rr = k - ci * KHW;
        int ky = rr / KW;
        int kx = rr - ky * KW;
        int yy = ho_s * stride + ky - pad;
        int xx = wo_s * stride + kx - pad;
        if (yy >= 0 && yy < H && xx >= 0 && xx < W)
          av = X[((size_t)(nb_s * Cin + ci) * H + yy) * W + xx];
      }
      As[sm * LDA + kk + sksel] = f2bf(av);
    }

#if HAVE_TDM
    if (wave == 0) __builtin_amdgcn_s_wait_tensorcnt(0);
#else
    if (k0 + BK < K) __builtin_prefetch(wrow_s + k0 + BK + skb, 0, 1);
#pragma unroll
    for (int e = 0; e < 8; ++e) {
      int k = k0 + skb + e;
      float bv = (nok_s && k < K) ? wrow_s[k] : 0.0f;
      Bsh[sn * LDB + skb + e] = f2bf(bv);
    }
#endif
    __syncthreads();

    // ---- fragments from LDS + 4 WMMAs ----
    FragU aF[2], bF[2];
#pragma unroll
    for (int s = 0; s < 2; ++s) {
      int mrow = waveM * 32 + s * 16 + lm;
      aF[s].q[0] = *(const float4*)(As + mrow * LDA + half * 16);
      aF[s].q[1] = *(const float4*)(As + mrow * LDA + half * 16 + 8);
      int nrow = waveN * 32 + s * 16 + lm;
#if HAVE_TDM
      Ld16U ld;
      const float* brow = Bf + nrow * LDBF + half * 16;
      ld.q[0] = *(const float4*)(brow);
      ld.q[1] = *(const float4*)(brow + 4);
      ld.q[2] = *(const float4*)(brow + 8);
      ld.q[3] = *(const float4*)(brow + 12);
#pragma unroll
      for (int h = 0; h < 16; ++h) bF[s].v[h] = f2bf(ld.f[h]);
#else
      bF[s].q[0] = *(const float4*)(Bsh + nrow * LDB + half * 16);
      bF[s].q[1] = *(const float4*)(Bsh + nrow * LDB + half * 16 + 8);
#endif
    }
#pragma unroll
    for (int i = 0; i < 2; ++i)
#pragma unroll
      for (int j = 0; j < 2; ++j)
        acc[i][j] = __builtin_amdgcn_wmma_f32_16x16x32_bf16(
            false, aF[i].v, false, bF[j].v, (short)0, acc[i][j], false, false);
    __syncthreads();
  }

  // ---- write back (NCHW scatter + bias) ----
#pragma unroll
  for (int i = 0; i < 2; ++i) {
#pragma unroll
    for (int r = 0; r < 8; ++r) {
      int m = m0 + waveM * 32 + i * 16 + r + half * 8;
      if (m < Mtot) {
        int wo = m % Wo; int tt = m / Wo; int ho = tt % Ho; int nb = tt / Ho;
#pragma unroll
        for (int j = 0; j < 2; ++j) {
          int n = n0 + waveN * 32 + j * 16 + lm;
          if (n < Cout)
            Y[((size_t)(nb * Cout + n) * Ho + ho) * Wo + wo] = acc[i][j][r] + Bs[n];
        }
      }
    }
  }
}

// ---------------------------------------------------------------------------
// Attention core for one batch image: S = softmax(Q^T K / sqrt(C)),
// out[c,i] = sum_j V[c,j] S[i,j]. One wave per 16 query rows; the 16xHW score
// strip lives in LDS between the two WMMA GEMM phases.
// ---------------------------------------------------------------------------
__global__ __launch_bounds__(32)
void vq_attn_core(const float* __restrict__ Q, const float* __restrict__ Kt,
                  const float* __restrict__ V, float* __restrict__ O,
                  int C, int HW, float scale)
{
  __shared__ float sS[16 * 256];   // HW <= 256 at the 16x16 attention stages
  const int lane = threadIdx.x;
  const int half = lane >> 4, lm = lane & 15;
  const int i0 = blockIdx.x * 16;
  const int b  = blockIdx.y;
  const float* qb = Q  + (size_t)b * C * HW;
  const float* kb = Kt + (size_t)b * C * HW;
  const float* vb = V  + (size_t)b * C * HW;
  float*       ob = O  + (size_t)b * C * HW;

  // Phase 1: scores S[i, j] = sum_c q[c,i] k[c,j]
  for (int j0 = 0; j0 < HW; j0 += 16) {
    v8f acc = {};
    for (int k0 = 0; k0 < C; k0 += 32) {
      v16bf a = {}, bf = {};
#pragma unroll
      for (int h = 0; h < 16; ++h) {
        int ka = k0 + half * 16 + h;
        a[h] = f2bf(qb[(size_t)ka * HW + i0 + lm]);
        int ki = k0 + half * 16 + h;
        bf[h] = f2bf(kb[(size_t)ki * HW + j0 + lm]);
      }
      acc = __builtin_amdgcn_wmma_f32_16x16x32_bf16(false, a, false, bf,
                                                    (short)0, acc, false, false);
    }
#pragma unroll
    for (int r = 0; r < 8; ++r)
      sS[(r + half * 8) * HW + j0 + lm] = acc[r] * scale;
  }
  __syncthreads();

  // Phase 2: row softmax (one lane per query row)
  if (lane < 16) {
    float* row = sS + lane * HW;
    float mx = -3.4e38f;
    for (int j = 0; j < HW; ++j) mx = fmaxf(mx, row[j]);
    float sum = 0.f;
    for (int j = 0; j < HW; ++j) { float e = __expf(row[j] - mx); row[j] = e; sum += e; }
    float inv = 1.0f / sum;
    for (int j = 0; j < HW; ++j) row[j] *= inv;
  }
  __syncthreads();

  // Phase 3: out[c, i] = sum_j S[i, j] v[c, j]   (A from LDS -> ds_load)
  for (int c0 = 0; c0 < C; c0 += 16) {
    v8f acc = {};
    for (int k0 = 0; k0 < HW; k0 += 32) {
      v16bf a = {}, bf = {};
#pragma unroll
      for (int h = 0; h < 16; ++h) {
        int ka = k0 + half * 16 + h;
        a[h] = f2bf(sS[lm * HW + ka]);
        int ki = k0 + half * 16 + h;
        bf[h] = f2bf(vb[(size_t)(c0 + lm) * HW + ki]);
      }
      acc = __builtin_amdgcn_wmma_f32_16x16x32_bf16(false, a, false, bf,
                                                    (short)0, acc, false, false);
    }
#pragma unroll
    for (int r = 0; r < 8; ++r)
      ob[(size_t)(c0 + lm) * HW + i0 + r + half * 8] = acc[r];
  }
}

// ---------------------------------------------------------------------------
// GroupNorm (+ optional swish). One block per (n, group).
// ---------------------------------------------------------------------------
__global__ __launch_bounds__(256)
void vq_groupnorm(const float* __restrict__ X, const float* __restrict__ G,
                  const float* __restrict__ Bt, float* __restrict__ Y,
                  int C, int HW, int groups, int do_swish)
{
  __shared__ float ssum[256], ssq[256];
  int n = blockIdx.x / groups, grp = blockIdx.x % groups;
  int cpg = C / groups;
  size_t base = ((size_t)n * C + (size_t)grp * cpg) * HW;
  int cnt = cpg * HW;
  float s = 0.f, q = 0.f;
  for (int i = threadIdx.x; i < cnt; i += 256) { float v = X[base + i]; s += v; q += v * v; }
  ssum[threadIdx.x] = s; ssq[threadIdx.x] = q;
  __syncthreads();
  for (int st = 128; st > 0; st >>= 1) {
    if ((int)threadIdx.x < st) {
      ssum[threadIdx.x] += ssum[threadIdx.x + st];
      ssq[threadIdx.x]  += ssq[threadIdx.x + st];
    }
    __syncthreads();
  }
  float mean = ssum[0] / cnt;
  float var  = ssq[0] / cnt - mean * mean;
  float inv  = rsqrtf(var + 1e-6f);
  for (int i = threadIdx.x; i < cnt; i += 256) {
    int c = grp * cpg + i / HW;
    float v = (X[base + i] - mean) * inv * G[c] + Bt[c];
    if (do_swish) v = v * (1.0f / (1.0f + __expf(-v)));
    Y[base + i] = v;
  }
}

__global__ __launch_bounds__(256)
void vq_add(const float* __restrict__ A, const float* __restrict__ B,
            float* __restrict__ Y, int n)
{
  int i = blockIdx.x * 256 + threadIdx.x;
  if (i < n) Y[i] = A[i] + B[i];
}

__global__ __launch_bounds__(256)
void vq_upsample2x(const float* __restrict__ X, float* __restrict__ Y,
                   int N, int C, int H, int W)
{
  int i = blockIdx.x * 256 + threadIdx.x;
  int W2 = 2 * W, H2 = 2 * H;
  int total = N * C * H2 * W2;
  if (i >= total) return;
  int x = i % W2; int t = i / W2; int y = t % H2; t /= H2; int c = t % C; int n = t / C;
  Y[i] = X[((size_t)(n * C + c) * H + (y >> 1)) * W + (x >> 1)];
}

// Codebook: nearest of 256 16-dim codes; writes zq (NCHW), idx (as float),
// accumulates sum of squared error for the loss.
__global__ __launch_bounds__(256)
void vq_codebook(const float* __restrict__ Z, const float* __restrict__ E,
                 float* __restrict__ ZQ, float* __restrict__ IDX,
                 float* __restrict__ lossAcc, int B_, int D, int H, int W)
{
  int p = blockIdx.x * 256 + threadIdx.x;
  int total = B_ * H * W;
  if (p >= total) return;
  int w = p % W; int t = p / W; int h = t % H; int b = t / H;
  float zv[16];
  for (int d = 0; d < D; ++d) zv[d] = Z[((size_t)(b * D + d) * H + h) * W + w];
  int best = 0; float bestd = 3.4e38f;
  for (int c = 0; c < 256; ++c) {
    float dd = 0.f;
    for (int d = 0; d < D; ++d) { float df = zv[d] - E[c * D + d]; dd += df * df; }
    if (dd < bestd) { bestd = dd; best = c; }
  }
  float err = 0.f;
  for (int d = 0; d < D; ++d) {
    float e = E[best * D + d];
    ZQ[((size_t)(b * D + d) * H + h) * W + w] = e;
    float df = e - zv[d]; err += df * df;
  }
  IDX[p] = (float)best;
  atomicAdd(lossAcc, err);
}

__global__ void vq_zero(float* p) { p[0] = 0.0f; }
__global__ void vq_finalize_loss(const float* acc, float* out, float count) {
  out[0] = 1.25f * acc[0] / count;   // mse + BETA*mse, forward value
}

// ===========================================================================
// Host side: rebuild the JAX pytree leaf order and walk the layer plan.
// ===========================================================================
namespace {

enum Kind { K_CONV = 0, K_DOWN, K_UP, K_RES, K_ATTN, K_GNSW };
struct ConvP { const float* w = nullptr; const float* b = nullptr; };
struct LayerP {
  Kind kind; int cin, cout;
  ConvP cv;                                    // conv/down/up
  const float *g1, *bg1, *g2, *bg2;            // res
  ConvP c1, c2, skip; bool has_skip;
  const float *g, *bg;                         // attn / gnsw
  ConvP q, k, v, o;                            // attn
};

void build_enc(std::vector<LayerP>& P) {
  auto add = [&](Kind k, int ci, int co) { LayerP l{}; l.kind = k; l.cin = ci; l.cout = co; P.push_back(l); };
  add(K_CONV, 1, 64);
  int chs[5] = {64, 64, 128, 128, 256}; int res = 128;
  for (int i = 0; i < 4; ++i) {
    int ci = chs[i], co = chs[i + 1];
    for (int r = 0; r < 2; ++r) { add(K_RES, ci, co); ci = co; if (res == 16) add(K_ATTN, ci, ci); }
    if (i != 3) { add(K_DOWN, co, co); res /= 2; }
  }
  add(K_RES, 256, 256); add(K_ATTN, 256, 256); add(K_RES, 256, 256);
  add(K_GNSW, 256, 256); add(K_CONV, 256, 16);
}

void build_dec(std::vector<LayerP>& P) {
  auto add = [&](Kind k, int ci, int co) { LayerP l{}; l.kind = k; l.cin = ci; l.cout = co; P.push_back(l); };
  add(K_CONV, 16, 256); add(K_RES, 256, 256); add(K_ATTN, 256, 256); add(K_RES, 256, 256);
  int chs[4] = {256, 128, 128, 64}; int cin = 256; int res = 16;
  for (int i = 0; i < 4; ++i) {
    int co = chs[i];
    for (int r = 0; r < 3; ++r) { add(K_RES, cin, co); cin = co; if (res == 16) add(K_ATTN, cin, cin); }
    if (i != 0) { add(K_UP, cin, cin); res *= 2; }
  }
  add(K_GNSW, cin, cin); add(K_CONV, cin, 1);
}

// Leaves per layer in JAX sorted-dict-key order.
void fill_params(std::vector<LayerP>& P, void* const* d_in, int& idx) {
  for (auto& l : P) {
    auto nx = [&]() { return (const float*)d_in[idx++]; };
    switch (l.kind) {
      case K_CONV: case K_DOWN: case K_UP:
        l.cv.b = nx(); l.cv.w = nx(); break;                        // b, w
      case K_RES:
        l.bg1 = nx(); l.bg2 = nx();                                 // bg1, bg2
        l.c1.b = nx(); l.c1.w = nx();                               // c1{b,w}
        l.c2.b = nx(); l.c2.w = nx();                               // c2{b,w}
        l.g1 = nx(); l.g2 = nx();                                   // g1, g2
        l.has_skip = (l.cin != l.cout);
        if (l.has_skip) { l.skip.b = nx(); l.skip.w = nx(); }       // skip{b,w}
        break;
      case K_ATTN:
        l.bg = nx(); l.g = nx();                                    // bg, g
        l.k.b = nx(); l.k.w = nx();                                 // k{b,w}
        l.o.b = nx(); l.o.w = nx();                                 // o{b,w}
        l.q.b = nx(); l.q.w = nx();                                 // q{b,w}
        l.v.b = nx(); l.v.w = nx();                                 // v{b,w}
        break;
      case K_GNSW:
        l.bg = nx(); l.g = nx(); break;                             // bg, g
    }
  }
}

struct Ctx {
  hipStream_t st;
  float *L0, *L1, *L2, *L3, *S0, *S1, *S2, *S3, *S4;
  int N;
};

void conv_launch(const Ctx& c, const float* X, const ConvP& p, float* Y,
                 int Cin, int H, int W, int Cout, int Ho, int Wo,
                 int Ksz, int stride, int pad) {
  int Mtot = c.N * Ho * Wo;
  dim3 g((Mtot + BM - 1) / BM, (Cout + BN - 1) / BN);
  vq_conv_wmma<<<g, 256, 0, c.st>>>(X, p.w, p.b, Y, c.N, Cin, H, W,
                                    Cout, Ho, Wo, Ksz, Ksz, stride, pad);
}

void gn_launch(const Ctx& c, const float* X, const float* g, const float* b,
               float* Y, int C, int H, int W, bool swish) {
  vq_groupnorm<<<c.N * 32, 256, 0, c.st>>>(X, g, b, Y, C, H * W, 32, swish ? 1 : 0);
}

void add_launch(const Ctx& c, const float* A, const float* B, float* Y, int n) {
  vq_add<<<(n + 255) / 256, 256, 0, c.st>>>(A, B, Y, n);
}

const float* run_plan(const Ctx& c, const std::vector<LayerP>& P,
                      const float* x0, int& H, float* final_out) {
  const float* cur = x0; int W = H;
  for (size_t li = 0; li < P.size(); ++li) {
    const LayerP& l = P[li];
    bool last = (li + 1 == P.size());
    float* out = (cur == c.L0) ? c.L1 : c.L0;
    if (last && final_out) out = final_out;
    switch (l.kind) {
      case K_CONV:
        conv_launch(c, cur, l.cv, out, l.cin, H, W, l.cout, H, W, 3, 1, 1);
        break;
      case K_DOWN: {
        // pad (0,1)x(0,1) + stride-2 conv == pad_top/left 0, OOB-bottom = 0
        int Ho = H / 2;
        conv_launch(c, cur, l.cv, out, l.cin, H, W, l.cout, Ho, Ho, 3, 2, 0);
        H = Ho; W = Ho;
        break;
      }
      case K_UP: {
        int H2 = 2 * H;
        int tot = c.N * l.cin * H2 * H2;
        vq_upsample2x<<<(tot + 255) / 256, 256, 0, c.st>>>(cur, c.L2, c.N, l.cin, H, W);
        conv_launch(c, c.L2, l.cv, out, l.cin, H2, H2, l.cout, H2, H2, 3, 1, 1);
        H = H2; W = H2;
        break;
      }
      case K_RES: {
        gn_launch(c, cur, l.g1, l.bg1, c.L2, l.cin, H, W, true);
        conv_launch(c, c.L2, l.c1, c.L3, l.cin, H, W, l.cout, H, W, 3, 1, 1);
        gn_launch(c, c.L3, l.g2, l.bg2, c.L2, l.cout, H, W, true);
        conv_launch(c, c.L2, l.c2, c.L3, l.cout, H, W, l.cout, H, W, 3, 1, 1);
        int n = c.N * l.cout * H * W;
        if (l.has_skip) {
          conv_launch(c, cur, l.skip, c.L2, l.cin, H, W, l.cout, H, W, 1, 1, 0);
          add_launch(c, c.L2, c.L3, out, n);
        } else {
          add_launch(c, cur, c.L3, out, n);
        }
        break;
      }
      case K_ATTN: {
        int C = l.cin, HW = H * W;
        gn_launch(c, cur, l.g, l.bg, c.S0, C, H, W, false);
        conv_launch(c, c.S0, l.q, c.S1, C, H, W, C, H, W, 1, 1, 0);
        conv_launch(c, c.S0, l.k, c.S2, C, H, W, C, H, W, 1, 1, 0);
        conv_launch(c, c.S0, l.v, c.S3, C, H, W, C, H, W, 1, 1, 0);
        vq_attn_core<<<dim3(HW / 16, c.N), 32, 0, c.st>>>(
            c.S1, c.S2, c.S3, c.S4, C, HW, 1.0f / sqrtf((float)C));
        conv_launch(c, c.S4, l.o, c.L2, C, H, W, C, H, W, 1, 1, 0);
        add_launch(c, cur, c.L2, out, c.N * C * HW);
        break;
      }
      case K_GNSW:
        gn_launch(c, cur, l.g, l.bg, out, l.cin, H, W, true);
        break;
    }
    cur = out;
  }
  return cur;
}

} // namespace

extern "C" void kernel_launch(void* const* d_in, const int* in_sizes, int n_in,
                              void* d_out, int out_size, void* d_ws, size_t ws_size,
                              hipStream_t stream) {
  (void)in_sizes; (void)n_in; (void)out_size;
  const int N = 8;
  const int DEC_ELEMS = 8 * 1 * 128 * 128;     // 131072
  const int IDX_ELEMS = 8 * 16 * 16;           // 2048

  // --- pytree leaf order: designs, codebook, decoder..., encoder...,
  //     post_quant{b,w}, quant{b,w} ---
  int idx = 0;
  const float* designs  = (const float*)d_in[idx++];
  const float* codebook = (const float*)d_in[idx++];

  std::vector<LayerP> dec, enc;
  build_dec(dec);
  build_enc(enc);
  fill_params(dec, d_in, idx);
  fill_params(enc, d_in, idx);
  ConvP post_quant; post_quant.b = (const float*)d_in[idx++]; post_quant.w = (const float*)d_in[idx++];
  ConvP quant;      quant.b      = (const float*)d_in[idx++]; quant.w      = (const float*)d_in[idx++];

  // --- workspace carve-up ---
  const size_t BIG = (size_t)8 * 64 * 128 * 128;   // largest activation (floats)
  const size_t SM  = (size_t)8 * 256 * 256;        // attention-stage tensors
  size_t need = (4 * BIG + 5 * SM + 16) * sizeof(float);
  if (ws_size < need) return;                       // deterministic guard
  float* ws = (float*)d_ws;
  Ctx c{stream,
        ws, ws + BIG, ws + 2 * BIG, ws + 3 * BIG,
        ws + 4 * BIG, ws + 4 * BIG + SM, ws + 4 * BIG + 2 * SM,
        ws + 4 * BIG + 3 * SM, ws + 4 * BIG + 4 * SM, N};
  float* lossAcc = ws + 4 * BIG + 5 * SM;
  float* outF = (float*)d_out;

  // --- encoder: [8,1,128,128] -> [8,16,16,16] ---
  int H = 128;
  const float* encOut = run_plan(c, enc, designs, H, nullptr);

  // --- quant 1x1 conv -> z ---
  conv_launch(c, encOut, quant, c.L2, 16, 16, 16, 16, 16, 16, 1, 1, 0);

  // --- codebook: zq (L3), idx, loss ---
  vq_zero<<<1, 1, 0, stream>>>(lossAcc);
  vq_codebook<<<(N * 16 * 16 + 255) / 256, 256, 0, stream>>>(
      c.L2, codebook, c.L3, outF + DEC_ELEMS, lossAcc, N, 16, 16, 16);
  vq_finalize_loss<<<1, 1, 0, stream>>>(lossAcc, outF + DEC_ELEMS + IDX_ELEMS,
                                        (float)(N * 16 * 16 * 16));

  // --- post_quant 1x1 conv ---
  conv_launch(c, c.L3, post_quant, c.L0, 16, 16, 16, 16, 16, 16, 1, 1, 0);

  // --- decoder: [8,16,16,16] -> decoded written straight into d_out ---
  H = 16;
  run_plan(c, dec, c.L0, H, outF);
}
